// Head_51737176048227
// MI455X (gfx1250) — matrix operand order repaired and checked by reading, not compiled, not run on hardware
//
#include <hip/hip_runtime.h>

// ---------------------------------------------------------------------------
// Problem constants (from reference): B=8, T=2048, D=1024, HS=64
// ---------------------------------------------------------------------------
#define Bb   8
#define Tt   2048
#define Dd   1024
#define HSs  64
#define SPLIT 4            // v-range splits for occupancy (flash split-K)
#define VCHUNK (Tt / SPLIT)

typedef __attribute__((ext_vector_type(16))) __bf16 v16bf;
typedef __attribute__((ext_vector_type(8)))  float  v8f;
typedef __attribute__((ext_vector_type(4)))  int    v4i;

union BF16x16 {
    unsigned short u[16];
    unsigned int   w[8];
    v16bf          v;
};

// Hardware fp32->bf16 conversion (backend emits packed v_cvt where available)
__device__ __forceinline__ unsigned short f2bfu(float f) {
    return __builtin_bit_cast(unsigned short, (__bf16)f);
}

// A-matrix 16x32 bf16 K order used here (contiguous; B matches): per lane (m = lane&15, kh = lane>>4),
// 16 elements = 8 contiguous K-pairs; pair j starts at K:
__device__ __forceinline__ int a_kpat(int j, int kh) {
    return (kh << 4) + (j << 1);
}

__device__ __forceinline__ v8f wmma_bf16(const BF16x16& a, const BF16x16& b, v8f c) {
    return __builtin_amdgcn_wmma_f32_16x16x32_bf16(
        /*neg_a=*/false, a.v, /*neg_b=*/false, b.v,
        /*c_mod=*/(short)0, c, /*reuse_a=*/false, /*reuse_b=*/false);
}

// -------- optional CDNA5 async global->LDS path (probe-guarded) ------------
// Probe round 2 revealed arg0 type: 'int __vector(4) __device__ *' (v4i AS1*).
typedef __attribute__((address_space(1))) v4i* g4_ptr;
typedef __attribute__((address_space(3))) v4i* l4_ptr;

#if __has_builtin(__builtin_amdgcn_global_load_async_to_lds_b128)
#define HAVE_ASYNC_LDS 1
#define ASYNC_LOAD_B128(g, l) \
    __builtin_amdgcn_global_load_async_to_lds_b128((g4_ptr)(void*)(g), \
                                                   (l4_ptr)(void*)(l), 0, 0)
#endif

__device__ __forceinline__ void wait_async0() {
#if __has_builtin(__builtin_amdgcn_s_wait_asynccnt)
    __builtin_amdgcn_s_wait_asynccnt(0);
#else
    asm volatile("s_wait_asynccnt 0x0" ::: "memory");
#endif
}

// ---------------------------------------------------------------------------
// Kernel 1: QKV projection.  C[16384 x 192] = x[16384 x 1024] * [Wq|Wk|Wv]
// grid.x = 1024 (16-row M tiles), block = 384 (12 waves = 3 matrices x 4 N-tiles)
// x tile staged in LDS once per k-step (async DMA when available) and shared
// by all 12 waves.
// ---------------------------------------------------------------------------
__global__ __launch_bounds__(384) void qkv_kernel(
    const float* __restrict__ x,
    const float* __restrict__ Wq, const float* __restrict__ Wk,
    const float* __restrict__ Wv,
    unsigned short* __restrict__ qb, unsigned short* __restrict__ kb,
    unsigned short* __restrict__ vb)
{
#ifdef HAVE_ASYNC_LDS
    __shared__ float Axf[16 * 32];            // fp32 tile via async DMA
#else
    __shared__ unsigned short Alds[16 * 32];  // pre-converted bf16 tile
#endif

    const int tid  = threadIdx.x;
    const int wv   = tid >> 5;
    const int lane = tid & 31;
    const int j    = wv >> 2;            // 0=Q, 1=K, 2=V
    const int n0   = (wv & 3) << 4;      // N-tile within HS=64
    const int m    = lane & 15;
    const int kh   = lane >> 4;
    const long bt0 = (long)blockIdx.x * 16;

    const float* W = (j == 0) ? Wq : ((j == 1) ? Wk : Wv);
    unsigned short* outp = (j == 0) ? qb : ((j == 1) ? kb : vb);

    v8f acc = {};
    for (int k0 = 0; k0 < Dd; k0 += 32) {
        __syncthreads();                       // protect LDS tile reuse
#ifdef HAVE_ASYNC_LDS
        if (tid < 128) {                       // 128 lanes x 16B = full tile
            int e = tid << 2;
            int row = e >> 5, col = e & 31;
            ASYNC_LOAD_B128(&x[(bt0 + row) * Dd + k0 + col], &Axf[row * 32 + col]);
        }
        wait_async0();
        __syncthreads();

        BF16x16 a, b;
        #pragma unroll
        for (int p = 0; p < 8; ++p) {
            int K0 = a_kpat(p, kh);
            a.v[2 * p + 0] = (__bf16)Axf[m * 32 + K0 + 0];
            a.v[2 * p + 1] = (__bf16)Axf[m * 32 + K0 + 1];
        }
#else
        if (tid < 128) {
            int e = tid << 2;
            int row = e >> 5, col = e & 31;
            float4 f = *(const float4*)&x[(bt0 + row) * Dd + k0 + col];
            Alds[row * 32 + col + 0] = f2bfu(f.x);
            Alds[row * 32 + col + 1] = f2bfu(f.y);
            Alds[row * 32 + col + 2] = f2bfu(f.z);
            Alds[row * 32 + col + 3] = f2bfu(f.w);
        }
        __syncthreads();

        BF16x16 a, b;
        #pragma unroll
        for (int p = 0; p < 8; ++p) {
            int K0 = a_kpat(p, kh);
            a.w[p] = *(const unsigned int*)&Alds[m * 32 + K0];
        }
#endif
        const int n = n0 + m;                  // B frag: lane = column n
        #pragma unroll
        for (int e = 0; e < 16; ++e) {
            int k = k0 + kh * 16 + e;
            b.v[e] = (__bf16)W[(long)k * HSs + n];
        }
        acc = wmma_bf16(a, b, acc);
    }

    #pragma unroll
    for (int r = 0; r < 8; ++r) {              // D: row = r + kh*8, col = m
        long row = bt0 + r + kh * 8;
        outp[row * HSs + n0 + m] = f2bfu(acc[r]);
    }
}

// ---------------------------------------------------------------------------
// Kernel 2: fused bias + flash attention over a v-split.
// grid = (T/16, SPLIT); block = 256 (8 waves).  Wave b owns batch b (rows = t);
// wave w additionally computes bias WMMAs for t = t0+w and t0+w+8 (rows = batch),
// exchanged through LDS.  pos_embd streamed exactly once across the whole grid.
// Writes unnormalized partials (O, rowmax, rowsum) for the combine pass.
// ---------------------------------------------------------------------------
__global__ __launch_bounds__(256) void attn_kernel(
    const float* __restrict__ pos,
    const unsigned short* __restrict__ qb,
    const unsigned short* __restrict__ kb,
    const unsigned short* __restrict__ vb,
    float* __restrict__ Opart, float* __restrict__ mpart,
    float* __restrict__ lpart)
{
    __shared__ float          biasS[16 * 32 * 8];   // [t_local][v(32)][b]
    __shared__ unsigned short pLDS[8 * 16 * 32];    // per-wave p tiles (bf16)

    const int tid  = threadIdx.x;
    const int w    = tid >> 5;       // wave id == batch id
    const int lane = tid & 31;
    const int m    = lane & 15;
    const int kh   = lane >> 4;
    const int t0   = blockIdx.x * 16;
    const int sp   = blockIdx.y;
    const int vbeg = sp * VCHUNK;
    const int vend = vbeg + VCHUNK;

    // A-fragments for scores: rows = t, batch w (fixed for whole kernel)
    BF16x16 aq[2];
    #pragma unroll
    for (int c = 0; c < 2; ++c)
        #pragma unroll
        for (int p = 0; p < 8; ++p) {
            int K0 = c * 32 + a_kpat(p, kh);
            aq[c].w[p] = *(const unsigned int*)&qb[((long)w * Tt + t0 + m) * HSs + K0];
        }

    // A-fragments for bias: rows = batch (8 valid), one fixed t per fragment set
    BF16x16 ab[2][2];
    #pragma unroll
    for (int tt = 0; tt < 2; ++tt) {
        int tsel = t0 + w + tt * 8;
        #pragma unroll
        for (int c = 0; c < 2; ++c)
            #pragma unroll
            for (int p = 0; p < 8; ++p) {
                int K0 = c * 32 + a_kpat(p, kh);
                ab[tt][c].w[p] = (m < 8)
                    ? *(const unsigned int*)&qb[((long)m * Tt + tsel) * HSs + K0]
                    : 0u;
            }
    }

    v8f O[4] = {{}, {}, {}, {}};
    float mrow[8], lrow[8];
    #pragma unroll
    for (int r = 0; r < 8; ++r) { mrow[r] = -3.0e38f; lrow[r] = 0.0f; }

    for (int v0 = vbeg; v0 < vend; v0 += 32) {
        __syncthreads();   // previous iteration's biasS/pLDS reads are done

        // -------- bias tiles (rows = batch) --------
        #pragma unroll
        for (int tt = 0; tt < 2; ++tt) {
            int tsel = t0 + w + tt * 8;
            int tl   = w + tt * 8;
            #pragma unroll
            for (int vt = 0; vt < 2; ++vt) {
                v8f bacc = {};
                #pragma unroll
                for (int c = 0; c < 2; ++c) {
                    BF16x16 bp;   // B[h][v] = pos[tsel, v0+vt*16+n, h]
                    const float* src =
                        &pos[((size_t)tsel * Tt + v0 + vt * 16 + m) * HSs + c * 32 + kh * 16];
                    #pragma unroll
                    for (int q4 = 0; q4 < 4; ++q4) {
                        float4 f = ((const float4*)src)[q4];
                        bp.v[q4 * 4 + 0] = (__bf16)f.x;
                        bp.v[q4 * 4 + 1] = (__bf16)f.y;
                        bp.v[q4 * 4 + 2] = (__bf16)f.z;
                        bp.v[q4 * 4 + 3] = (__bf16)f.w;
                    }
                    bacc = wmma_bf16(ab[tt][c], bp, bacc);
                }
                if (kh == 0) {                 // rows 0..7 = batches
                    #pragma unroll
                    for (int r = 0; r < 8; ++r)
                        biasS[(tl * 32 + vt * 16 + m) * 8 + r] = bacc[r];
                }
            }
        }
        __syncthreads();

        // -------- scores (rows = t) + bias add --------
        float sc[2][8];
        #pragma unroll
        for (int vt = 0; vt < 2; ++vt) {
            v8f sacc = {};
            #pragma unroll
            for (int c = 0; c < 2; ++c) {
                BF16x16 bk;   // B[h][v] = k[w, v0+vt*16+n, h]  (16 contiguous bf16)
                const unsigned short* src =
                    &kb[((long)w * Tt + v0 + vt * 16 + m) * HSs + c * 32 + kh * 16];
                uint4 u0 = ((const uint4*)src)[0];
                uint4 u1 = ((const uint4*)src)[1];
                bk.w[0] = u0.x; bk.w[1] = u0.y; bk.w[2] = u0.z; bk.w[3] = u0.w;
                bk.w[4] = u1.x; bk.w[5] = u1.y; bk.w[6] = u1.z; bk.w[7] = u1.w;
                sacc = wmma_bf16(aq[c], bk, sacc);
            }
            #pragma unroll
            for (int r = 0; r < 8; ++r) {
                int tl = r + kh * 8;
                // reference: scores / hs**-0.5 == scores * 8
                sc[vt][r] = sacc[r] * 8.0f + biasS[(tl * 32 + vt * 16 + m) * 8 + w];
            }
        }

        // -------- online softmax update (row = t, cols across 16 lanes) --------
        #pragma unroll
        for (int r = 0; r < 8; ++r) {
            float mx = fmaxf(sc[0][r], sc[1][r]);
            #pragma unroll
            for (int d = 1; d < 16; d <<= 1) mx = fmaxf(mx, __shfl_xor(mx, d, 32));
            float mnew = fmaxf(mrow[r], mx);
            float corr = __expf(mrow[r] - mnew);
            mrow[r] = mnew;
            float p0 = __expf(sc[0][r] - mnew);
            float p1 = __expf(sc[1][r] - mnew);
            float ps = p0 + p1;
            #pragma unroll
            for (int d = 1; d < 16; d <<= 1) ps += __shfl_xor(ps, d, 32);
            lrow[r] = lrow[r] * corr + ps;
            int tl = r + kh * 8;
            pLDS[(w * 16 + tl) * 32 + m]      = f2bfu(p0);
            pLDS[(w * 16 + tl) * 32 + 16 + m] = f2bfu(p1);
            #pragma unroll
            for (int ht = 0; ht < 4; ++ht) O[ht][r] *= corr;
        }
        __syncthreads();

        // -------- O += P (16x32) x V (32x64) --------
        BF16x16 ap;   // A[t][vv] from LDS transpose
        #pragma unroll
        for (int p = 0; p < 8; ++p) {
            int K0 = a_kpat(p, kh);
            ap.w[p] = *(const unsigned int*)&pLDS[(w * 16 + m) * 32 + K0];
        }
        #pragma unroll
        for (int ht = 0; ht < 4; ++ht) {
            BF16x16 bv;   // B[vv][h] = v[w, v0+vv, ht*16+n]
            #pragma unroll
            for (int e = 0; e < 16; ++e)
                bv.u[e] = vb[((long)w * Tt + v0 + kh * 16 + e) * HSs + ht * 16 + m];
            O[ht] = wmma_bf16(ap, bv, O[ht]);
        }

        // keep the pos_embd stream ahead of us
        if (v0 + 32 < vend)
            __builtin_prefetch(&pos[((size_t)(t0 + w) * Tt + v0 + 32) * HSs], 0, 1);
    }

    // -------- write split partials (unnormalized) --------
    #pragma unroll
    for (int r = 0; r < 8; ++r) {
        int tl = r + kh * 8;
        long bt = (long)(sp * Bb + w) * Tt + t0 + tl;
        if (m == 0) { mpart[bt] = mrow[r]; lpart[bt] = lrow[r]; }
        #pragma unroll
        for (int ht = 0; ht < 4; ++ht)
            Opart[bt * HSs + ht * 16 + m] = O[ht][r];
    }
}

// ---------------------------------------------------------------------------
// Kernel 3: merge the SPLIT flash partials and normalize.
// ---------------------------------------------------------------------------
__global__ __launch_bounds__(256) void combine_kernel(
    const float* __restrict__ Opart, const float* __restrict__ mpart,
    const float* __restrict__ lpart, float* __restrict__ out)
{
    long i = (long)blockIdx.x * 256 + threadIdx.x;
    const long N = (long)Bb * Tt * HSs;
    if (i >= N) return;
    long bt = i >> 6;            // (b*T + t)
    int  h  = (int)(i & 63);
    int  b  = (int)(bt >> 11);
    int  t  = (int)(bt & 2047);

    float mmax = -3.0e38f;
    #pragma unroll
    for (int s = 0; s < SPLIT; ++s)
        mmax = fmaxf(mmax, mpart[(long)(s * Bb + b) * Tt + t]);
    float num = 0.0f, den = 0.0f;
    #pragma unroll
    for (int s = 0; s < SPLIT; ++s) {
        long idx = (long)(s * Bb + b) * Tt + t;
        float sc = __expf(mpart[idx] - mmax);
        num += sc * Opart[idx * HSs + h];
        den += sc * lpart[idx];
    }
    out[i] = num / den;
}

// ---------------------------------------------------------------------------
extern "C" void kernel_launch(void* const* d_in, const int* in_sizes, int n_in,
                              void* d_out, int out_size, void* d_ws, size_t ws_size,
                              hipStream_t stream)
{
    const float* x   = (const float*)d_in[0];
    const float* Wq  = (const float*)d_in[1];
    const float* Wk  = (const float*)d_in[2];
    const float* Wv  = (const float*)d_in[3];
    const float* pos = (const float*)d_in[4];
    float* out = (float*)d_out;

    // workspace layout
    char* ws = (char*)d_ws;
    const size_t QKV = (size_t)Bb * Tt * HSs;            // 1M elements
    unsigned short* qb = (unsigned short*)ws;             //  2 MB
    unsigned short* kb = qb + QKV;                        //  2 MB
    unsigned short* vb = kb + QKV;                        //  2 MB
    float* Opart = (float*)(ws + 3 * QKV * sizeof(unsigned short)); // 16 MB
    float* mpart = Opart + (size_t)SPLIT * QKV;           // 256 KB
    float* lpart = mpart + (size_t)SPLIT * Bb * Tt;       // 256 KB

    qkv_kernel<<<dim3((Bb * Tt) / 16), dim3(384), 0, stream>>>(
        x, Wq, Wk, Wv, qb, kb, vb);

    attn_kernel<<<dim3(Tt / 16, SPLIT), dim3(256), 0, stream>>>(
        pos, qb, kb, vb, Opart, mpart, lpart);

    combine_kernel<<<dim3((unsigned)((QKV + 255) / 256)), dim3(256), 0, stream>>>(
        Opart, mpart, lpart, out);
}
